// EdgeSTGUBlock_16320875724949
// MI455X (gfx1250) — compile-verified, matrix operand order checked by reading, not run on hardware
//
#include <hip/hip_runtime.h>

// ---------------------------------------------------------------------------
// Fused EdgeSTGU block for gfx1250 (MI455X), wave32 + WMMA f16.
//   out = h + scatter_add_dst( sigmoid(gelu([src|dst]@W1+b1)@W2+b2) * (src@Wv+bv) )
// with src/dst = LayerNorm(h) gathered per edge.
//
// Kernel 1 (pack_weights): repack Wv/W1 (f32 row-major) into fragment-major
//   f16 in d_ws, in the exact per-lane VGPR order of the WMMA B operand, so
//   the hot loop loads each B fragment as two coalesced global_load_b128.
// Kernel 2 (edge_stgu_fused): one workgroup per batch; LN + both GEMMs +
//   gate + scatter all fused; intermediates live in LDS only.
// ---------------------------------------------------------------------------

#define B_TOTAL   8192
#define NN        21      // nodes
#define D         256     // feature dim
#define GH        512     // gate hidden
#define E_EDGES   40
#define E_PAD     48      // 3 WMMA M-tiles of 16
#define XS_STRIDE 264     // padded f16 row stride (528 B, 16B aligned)
#define LN_EPS    1e-5f

#define W1_FRAGS  (32 * 16)             // nt-tiles x k-steps
#define WV_FRAGS  (16 * 8)
#define W1_PACK_HALFS ((size_t)W1_FRAGS * 32 * 16)   // 256 Ki halfs = 512 KB
#define WV_PACK_HALFS ((size_t)WV_FRAGS * 32 * 16)   //  64 Ki halfs = 128 KB

typedef __attribute__((ext_vector_type(16))) _Float16 v16h;
typedef __attribute__((ext_vector_type(8)))  _Float16 v8h;
typedef __attribute__((ext_vector_type(8)))  float    v8f;

// K packing of one lane's 16 halfs for the 16x32 f16 A/B operand
// (ISA 7.12.2): pair j -> k = 2j + (j>=4?8:0) + hi*8.
__device__ __forceinline__ int k_of_pair(int j, int hi) {
    return 2 * j + ((j >= 4) ? 8 : 0) + hi * 8;
}

// A fragment: lane L holds row M=L%16; its 16 halfs are two 16B-contiguous
// K-chunks at +hi*8 and +16+hi*8 -> two ds_load_b128 from the LDS x rows.
__device__ __forceinline__ v16h load_a_chunks(const _Float16* row, int hi) {
    v8h lo = *(const v8h*)(row + hi * 8);
    v8h hb = *(const v8h*)(row + 16 + hi * 8);
    return __builtin_shufflevector(lo, hb, 0,1,2,3,4,5,6,7,8,9,10,11,12,13,14,15);
}

// B fragment from the pre-packed f16 image: lane L reads its own 32 B chunk
// (two global_load_b128, fully coalesced: 1 KB per wave per fragment).
__device__ __forceinline__ v16h load_b_packed(const _Float16* __restrict__ Wp,
                                              int frag, int lane) {
    const v8h* p = (const v8h*)(Wp + ((size_t)frag * 32 + lane) * 16);
    v8h lo = p[0];
    v8h hb = p[1];
    return __builtin_shufflevector(lo, hb, 0,1,2,3,4,5,6,7,8,9,10,11,12,13,14,15);
}

// ---- prep: f32 row-major W[K][N]  ->  fragment-major f16 ----
__global__ __launch_bounds__(256)
void pack_weights(const float* __restrict__ Wv, const float* __restrict__ W1,
                  _Float16* __restrict__ w1p, _Float16* __restrict__ wvp) {
    int id = blockIdx.x * 256 + threadIdx.x;
    const float* W;
    _Float16* P;
    int ldN, kmax_frag;
    if (id < W1_FRAGS * 32) { W = W1; P = w1p; ldN = GH; kmax_frag = 16; }
    else {
        id -= W1_FRAGS * 32;
        if (id >= WV_FRAGS * 32) return;
        W = Wv; P = wvp; ldN = D; kmax_frag = 8;
    }
    const int lane = id & 31;
    const int frag = id >> 5;
    const int k    = frag % kmax_frag;      // K-step
    const int nt   = frag / kmax_frag;      // N-tile
    const int lm   = lane & 15;
    const int hi   = lane >> 4;
    const int nG   = nt * 16 + lm;
    const int kb   = k * 32;
    _Float16* o = P + ((size_t)frag * 32 + lane) * 16;
    #pragma unroll
    for (int j = 0; j < 8; ++j) {
        const int kl = k_of_pair(j, hi);
        o[2 * j]     = (_Float16)W[(size_t)(kb + kl)     * ldN + nG];
        o[2 * j + 1] = (_Float16)W[(size_t)(kb + kl + 1) * ldN + nG];
    }
}

// ---- main fused kernel: one workgroup (8 wave32) per batch ----
__global__ __launch_bounds__(256)
void edge_stgu_fused(const float* __restrict__ h,
                     const int*   __restrict__ src,
                     const int*   __restrict__ dst,
                     const float* __restrict__ ln_w,
                     const float* __restrict__ ln_b,
                     const _Float16* __restrict__ w1p,  // packed W1
                     const _Float16* __restrict__ wvp,  // packed Wv
                     const float* __restrict__ bv,
                     const float* __restrict__ b1,
                     const float* __restrict__ W2,
                     const float* __restrict__ b2,
                     float*       __restrict__ out) {
    __shared__ float    hbuf[NN * D];            // residual + LN input (21.5 KB)
    __shared__ _Float16 xs[NN][XS_STRIDE];       // LN output, f16      (10.9 KB)
    __shared__ float    agg[NN * D];             // scatter accumulator (21.5 KB)
    __shared__ float    gatesum[E_PAD];
    __shared__ int      sidx[E_PAD], didx[E_PAD];

    const int b    = blockIdx.x;
    const int tid  = threadIdx.x;
    const int lane = tid & 31;
    const int wave = tid >> 5;
    const int lm   = lane & 15;   // M row (A) or N col (B/C) within a tile
    const int hi   = lane >> 4;   // which K/M half this lane covers

    if (tid < E_PAD) {
        // pad with node 0: those rows read valid data and are masked in the
        // epilogues, so the GEMM loops run guard-free with full EXEC.
        sidx[tid]    = (tid < E_EDGES) ? src[tid] : 0;
        didx[tid]    = (tid < E_EDGES) ? dst[tid] : 0;
        gatesum[tid] = 0.0f;
    }

    // ---- stage h[b] into LDS, zero the aggregator (coalesced b128) ----
    const float4* hg  = (const float4*)(h + (size_t)b * NN * D);
    float4*       hb4 = (float4*)hbuf;
    float4*       ag4 = (float4*)agg;
    const float4  z4  = make_float4(0.f, 0.f, 0.f, 0.f);
    for (int i = tid; i < NN * D / 4; i += 256) { hb4[i] = hg[i]; ag4[i] = z4; }
    __syncthreads();

    // ---- LayerNorm: one wave per node row, wave32 shfl reductions ----
    for (int r = wave; r < NN; r += 8) {
        float v[8];
        float s = 0.f;
        #pragma unroll
        for (int j = 0; j < 8; ++j) { v[j] = hbuf[r * D + lane * 8 + j]; s += v[j]; }
        #pragma unroll
        for (int m = 16; m >= 1; m >>= 1) s += __shfl_xor(s, m, 32);
        const float mu = s * (1.0f / D);
        float q = 0.f;
        #pragma unroll
        for (int j = 0; j < 8; ++j) { float t = v[j] - mu; q += t * t; }
        #pragma unroll
        for (int m = 16; m >= 1; m >>= 1) q += __shfl_xor(q, m, 32);
        const float rs = rsqrtf(q * (1.0f / D) + LN_EPS);
        #pragma unroll
        for (int j = 0; j < 8; ++j) {
            const int c = lane * 8 + j;
            xs[r][c] = (_Float16)((v[j] - mu) * rs * ln_w[c] + ln_b[c]);
        }
    }
    __syncthreads();

    // ---- GEMM2: [48 x 512] x W1[512 x 512], fused GeLU + W2-dot epilogue ----
    // 3 M-tiles x 32 N-tiles = 96 tiles round-robined over 8 waves.
    for (int t = wave; t < 96; t += 8) {
        const int mt = t >> 5;          // 0..2
        const int nt = t & 31;          // 0..31
        const int e  = mt * 16 + lm;    // this lane's A row (edge id)
        const int nG = nt * 16 + lm;    // this lane's B/C column
        // hoist gather addresses out of the K loop
        const _Float16* srow = &xs[sidx[e]][0];
        const _Float16* drow = &xs[didx[e]][0];
        // pull this tile's packed-W1 stream toward the caches
        __builtin_prefetch(w1p + ((size_t)nt * 16 * 32) * 16, 0, 1);
        v8f acc = {};
        #pragma unroll 4
        for (int k = 0; k < 16; ++k) {
            const int cbase = k * 32;   // column in the [src|dst] concat
            const v16h a  = load_a_chunks((cbase < D) ? (srow + cbase)
                                                      : (drow + cbase - D), hi);
            const v16h bf = load_b_packed(w1p, nt * 16 + k, lane);
            acc = __builtin_amdgcn_wmma_f32_16x16x32_f16(
                      false, a, false, bf, (short)0, acc, false, false);
        }
        // epilogue: gelu(acc + b1) * W2[n], reduce across the 16 lanes that
        // share an edge row, one ds_add_f32 per row per half.
        const float b1v = b1[nG];
        const float w2v = W2[nG];
        #pragma unroll
        for (int i = 0; i < 8; ++i) {
            const int e2 = mt * 16 + i + hi * 8;
            const float x  = acc[i] + b1v;
            const float gl = 0.5f * x * (1.0f + erff(x * 0.70710678118f));
            float contrib = (e2 < E_EDGES) ? gl * w2v : 0.0f;
            #pragma unroll
            for (int m = 8; m >= 1; m >>= 1) contrib += __shfl_xor(contrib, m, 32);
            if (lm == 0 && e2 < E_EDGES) atomicAdd(&gatesum[e2], contrib);
        }
    }
    __syncthreads();

    // ---- gates = sigmoid(. + b2) ----
    if (tid < E_EDGES)
        gatesum[tid] = 1.0f / (1.0f + expf(-(gatesum[tid] + b2[0])));
    __syncthreads();

    // ---- GEMM1: values = src[48 x 256] x Wv[256 x 256], fused gate*scatter ----
    for (int t = wave; t < 48; t += 8) {
        const int mt = t >> 4;          // 0..2
        const int nt = t & 15;          // 0..15
        const int e  = mt * 16 + lm;
        const int nG = nt * 16 + lm;
        const _Float16* srow = &xs[sidx[e]][0];
        v8f acc = {};
        #pragma unroll 4
        for (int k = 0; k < 8; ++k) {
            const v16h a  = load_a_chunks(srow + k * 32, hi);
            const v16h bf = load_b_packed(wvp, nt * 8 + k, lane);
            acc = __builtin_amdgcn_wmma_f32_16x16x32_f16(
                      false, a, false, bf, (short)0, acc, false, false);
        }
        const float bvv = bv[nG];
        #pragma unroll
        for (int i = 0; i < 8; ++i) {
            const int e2 = mt * 16 + i + hi * 8;
            if (e2 < E_EDGES) {
                const float msg = gatesum[e2] * (acc[i] + bvv);
                atomicAdd(&agg[didx[e2] * D + nG], msg);   // ds_add_f32
            }
        }
    }
    __syncthreads();

    // ---- out = residual + aggregated (coalesced b128 stores) ----
    float4* og4 = (float4*)(out + (size_t)b * NN * D);
    for (int i = tid; i < NN * D / 4; i += 256) {
        const float4 hv = hb4[i];
        const float4 av = ag4[i];
        og4[i] = make_float4(hv.x + av.x, hv.y + av.y, hv.z + av.z, hv.w + av.w);
    }
}

extern "C" void kernel_launch(void* const* d_in, const int* in_sizes, int n_in,
                              void* d_out, int out_size, void* d_ws, size_t ws_size,
                              hipStream_t stream) {
    const float* h    = (const float*)d_in[0];
    const int*   src  = (const int*)  d_in[1];
    const int*   dst  = (const int*)  d_in[2];
    const float* ln_w = (const float*)d_in[3];
    const float* ln_b = (const float*)d_in[4];
    const float* Wv   = (const float*)d_in[5];
    const float* bv   = (const float*)d_in[6];
    const float* W1   = (const float*)d_in[7];
    const float* b1   = (const float*)d_in[8];
    const float* W2   = (const float*)d_in[9];
    const float* b2   = (const float*)d_in[10];
    float* out = (float*)d_out;

    _Float16* w1p = (_Float16*)d_ws;                       // 512 KB
    _Float16* wvp = (_Float16*)d_ws + W1_PACK_HALFS;       // 128 KB

    // 1) repack weights into WMMA-B fragment order (f16), every call.
    const int pack_items  = (W1_FRAGS + WV_FRAGS) * 32;    // 20480
    const int pack_blocks = (pack_items + 255) / 256;
    hipLaunchKernelGGL(pack_weights, dim3(pack_blocks), dim3(256), 0, stream,
                       Wv, W1, w1p, wvp);

    // 2) fused block, one workgroup per batch.
    hipLaunchKernelGGL(edge_stgu_fused, dim3(B_TOTAL), dim3(256), 0, stream,
                       h, src, dst, ln_w, ln_b, w1p, wvp, bv, b1, W2, b2, out);
}
